// MLPredictor_7919919693879
// MI455X (gfx1250) — matrix operand order repaired and checked by reading, not compiled, not run on hardware
//
#include <hip/hip_runtime.h>
#include <hip/hip_bf16.h>

// CDNA5 (gfx1250) fused edge-MLP:  score[e] = relu(concat(h[src],h[dst]) @ W1 + b1) @ W2 + b2
// bf16 WMMA (16x16x32, f32 accumulate), hidden activations never touch memory.
// Round 2: 8 waves / 256 threads per block, 128 edges/block, ~297KB LDS.
// Rationale: round-1's 165KB block still capped the WGP at one block (2*165KB > 320KB),
// i.e. 1 wave/SIMD. One big block at 297KB gives 2 waves/SIMD for the same DS-per-WMMA,
// which matters because the kernel is LDS-bandwidth/latency bound around the WMMA pipe.

typedef __attribute__((ext_vector_type(16))) __bf16 bf16x16;
typedef __attribute__((ext_vector_type(8)))  __bf16 bf16x8;
typedef __attribute__((ext_vector_type(4)))  __bf16 bf16x4;
typedef __attribute__((ext_vector_type(8)))  float  f32x8;

#define HFEAT    512
#define KTOT     1024
#define ASTRIDE  1032          // bf16 elements per LDS row (1024 + 16B pad -> conflict-free b128)
#define NWAVES   8
#define EPB      (NWAVES * 16) // 128 edges per block

// ---- prep: W1 [K=1024][N=512] fp32  ->  W1t [N=512][K=1024] bf16 (workspace) ----
__global__ __launch_bounds__(256) void w1_transpose_bf16(const float* __restrict__ W1,
                                                         __bf16* __restrict__ W1t) {
  int idx = blockIdx.x * 256 + threadIdx.x;     // 524288 elements
  int k = idx >> 9;                             // coalesced read of W1 row k
  int n = idx & 511;
  W1t[((size_t)n << 10) + k] = (__bf16)W1[idx];
}

__device__ __forceinline__ void cvt_store4(__bf16* p, float4 v) {
  bf16x4 t;
  t.x = (__bf16)v.x; t.y = (__bf16)v.y; t.z = (__bf16)v.z; t.w = (__bf16)v.w;
  *(bf16x4*)p = t;   // 8B LDS store
}

__global__ __launch_bounds__(256) void edge_mlp_wmma(
    const float*      __restrict__ h,
    const long long*  __restrict__ src,
    const long long*  __restrict__ dst,
    const float*      __restrict__ b1,
    const float*      __restrict__ W2,
    const float*      __restrict__ b2,
    const __bf16*     __restrict__ W1t,
    float*            __restrict__ out,
    int nEdges)
{
  __shared__ __bf16 As[EPB * ASTRIDE];   // 128 gathered+concat'd edge rows, bf16 (~264KB)
  __shared__ __bf16 Bs[16  * ASTRIDE];   // one 16-column W1 panel, K-major       (~33KB)

  const int tid  = threadIdx.x;
  const int wid  = tid >> 5;
  const int lane = tid & 31;
  const int half = lane >> 4;            // 0: lanes 0-15, 1: lanes 16-31
  const int l15  = lane & 15;
  const int edgeBase = blockIdx.x * EPB + wid * 16;

  // ---- stage A: gather h[src]/h[dst], convert fp32->bf16, one 1024-wide row per edge ----
  for (int e = 0; e < 16; ++e) {
    const int edge = edgeBase + e;
    __bf16* arow = As + (wid * 16 + e) * ASTRIDE;
    if (edge < nEdges) {
      const float4* ps = (const float4*)(h + (size_t)src[edge] * HFEAT);
      const float4* pd = (const float4*)(h + (size_t)dst[edge] * HFEAT);
      #pragma unroll
      for (int it = 0; it < 4; ++it) {
        int c4 = it * 32 + lane;                       // float4 index within 512
        cvt_store4(arow + c4 * 4,         ps[c4]);     // src half  [0,512)
        cvt_store4(arow + 512 + c4 * 4,   pd[c4]);     // dst half  [512,1024)
      }
    } else {
      bf16x4 z = {(__bf16)0.f, (__bf16)0.f, (__bf16)0.f, (__bf16)0.f};
      #pragma unroll
      for (int it = 0; it < 8; ++it) {
        int c4 = it * 32 + lane;
        *(bf16x4*)(arow + c4 * 4) = z;
      }
    }
  }

  float sAcc[8] = {0.f,0.f,0.f,0.f,0.f,0.f,0.f,0.f};
  // A fragment base: row = this wave's edge (lane&15); K chunks at {0,16}+half*8
  const __bf16* aBase = As + (wid * 16 + l15) * ASTRIDE + half * 8;

  for (int nb = 0; nb < 32; ++nb) {
    const int n0 = nb * 16;

    __syncthreads();
    // cooperative B panel load: columns n0..n0+15, all K (contiguous rows of W1t)
    for (int i = tid; i < 16 * 128; i += 256) {
      int r = i >> 7;          // column 0..15
      int c = i & 127;         // 16B chunk within 2KB row
      *(bf16x8*)(Bs + r * ASTRIDE + c * 8) =
          *(const bf16x8*)(W1t + ((size_t)(n0 + r) << 10) + c * 8);
    }
    __syncthreads();

    f32x8 acc = {0.f,0.f,0.f,0.f,0.f,0.f,0.f,0.f};
    // B fragment base: column = lane&15; K chunks at {0,16}+half*8
    const __bf16* bBase = Bs + l15 * ASTRIDE + half * 8;

    #pragma unroll 4
    for (int kb = 0; kb < KTOT; kb += 32) {
      bf16x8 a0  = *(const bf16x8*)(aBase + kb);        // K = kb + half*8 + [0,8)
      bf16x8 a1  = *(const bf16x8*)(aBase + kb + 16);   // K = kb + half*8 + 16 + [0,8)
      bf16x8 bb0 = *(const bf16x8*)(bBase + kb);        // K = kb + half*8 + [0,8)
      bf16x8 bb1 = *(const bf16x8*)(bBase + kb + 16);   // K = kb + half*8 + 16 + [0,8)
      bf16x16 a, b;
      #pragma unroll
      for (int i = 0; i < 8; ++i) {
        a[i] = a0[i]; a[i + 8] = a1[i];
        b[i] = bb0[i]; b[i + 8] = bb1[i];
      }
      acc = __builtin_amdgcn_wmma_f32_16x16x32_bf16(false, a, false, b,
                                                    (short)0, acc, false, false);
    }

    // fused epilogue for this 16-column chunk: bias + ReLU + dot with W2 slice
    const float b1v = b1[n0 + l15];
    const float w2v = W2[n0 + l15];
    #pragma unroll
    for (int r = 0; r < 8; ++r) {
      float hv = acc[r] + b1v;
      hv = hv > 0.f ? hv : 0.f;
      sAcc[r] += hv * w2v;
    }
  }

  // reduce over the 16 lanes of each half (row M = half*8 + r lives in that half)
  #pragma unroll
  for (int r = 0; r < 8; ++r) {
    float v = sAcc[r];
    v += __shfl_xor(v, 1, 32);
    v += __shfl_xor(v, 2, 32);
    v += __shfl_xor(v, 4, 32);
    v += __shfl_xor(v, 8, 32);
    sAcc[r] = v;
  }
  if (l15 == 0) {
    const float bias2 = b2[0];
    #pragma unroll
    for (int r = 0; r < 8; ++r) {
      int edge = edgeBase + half * 8 + r;
      if (edge < nEdges) out[edge] = sAcc[r] + bias2;
    }
  }
}

extern "C" void kernel_launch(void* const* d_in, const int* in_sizes, int n_in,
                              void* d_out, int out_size, void* d_ws, size_t ws_size,
                              hipStream_t stream) {
  const float*     h_  = (const float*)d_in[0];
  const long long* src = (const long long*)d_in[1];   // reference dtype: int64
  const long long* dst = (const long long*)d_in[2];
  const float*     W1  = (const float*)d_in[3];
  const float*     b1  = (const float*)d_in[4];
  const float*     W2  = (const float*)d_in[5];
  const float*     b2  = (const float*)d_in[6];
  float*           out = (float*)d_out;
  __bf16*          W1t = (__bf16*)d_ws;               // 512*1024*2 = 1 MB scratch

  const int nEdges = in_sizes[1];

  // 1) W1 -> bf16, pre-transposed to [N][K] so B fragments are K-contiguous
  w1_transpose_bf16<<<(KTOT * HFEAT) / 256, 256, 0, stream>>>(W1, W1t);

  // 2) fused gather + GEMM(WMMA bf16) + bias + ReLU + W2-dot
  const int blocks = (nEdges + EPB - 1) / EPB;        // 1250 for E=160000
  edge_mlp_wmma<<<blocks, 256, 0, stream>>>(h_, src, dst, b1, W2, b2, W1t, out, nEdges);
}
